// SelectedUnitsHead_2534030705150
// MI455X (gfx1250) — matrix operand order repaired and checked by reading, not compiled, not run on hardware
//
#include <hip/hip_runtime.h>
#include <hip/hip_bf16.h>

#define B_     256
#define N_     512
#define S_     32
#define D_ENT  256
#define D_KEY  32
#define D_IN   1024
#define D_FUNC 256
#define D_TYPE 259
#define H_     32
#define NP1    513   // N + 1 (end flag)
#define SP1    33    // S + 1

typedef __attribute__((ext_vector_type(16))) __bf16 v16bf;
typedef __attribute__((ext_vector_type(8)))  float  v8f;
typedef __attribute__((ext_vector_type(4)))  unsigned int v4u;
typedef __attribute__((ext_vector_type(4)))  int    v4i;
typedef __attribute__((ext_vector_type(8)))  int    v8i;

// ---------- scalar helpers ----------
__device__ __forceinline__ __bf16 b16(float f) {
  unsigned u = __builtin_bit_cast(unsigned, f);
  return __builtin_bit_cast(__bf16, (unsigned short)(u >> 16));
}
__device__ __forceinline__ float sigf(float x) { return 1.0f / (1.0f + expf(-x)); }

// ---------- WMMA ----------
__device__ __forceinline__ v8f wmma_bf16(v16bf a, v16bf b, v8f c) {
  return __builtin_amdgcn_wmma_f32_16x16x32_bf16(false, a, false, b, (short)0, c, false, false);
}

// A-matrix 16x32 bf16 fragment: lane holds row M=lane&15,
// K pattern = kh+{0..15} (contiguous 16), kh = k0 + 16*(lane>>4).
__device__ __forceinline__ v16bf fragA_g32(const float* m, int stride, int row,
                                           int k0, int klim, int lane) {
  const float* p = m + (size_t)row * stride;
  int kh = k0 + ((lane >> 4) << 4);
  v16bf a;
#pragma unroll
  for (int i = 0; i < 8; ++i) {
    int ka = kh + i, kb = kh + 8 + i;
    a[i]     = b16(ka < klim ? p[ka] : 0.0f);
    a[i + 8] = b16(kb < klim ? p[kb] : 0.0f);
  }
  return a;
}
__device__ __forceinline__ v16bf fragA_lds(const __bf16* m, int stride, int row,
                                           int k0, int lane) {
  const __bf16* p = m + (size_t)row * stride;
  int kh = k0 + ((lane >> 4) << 4);
  v16bf a;
#pragma unroll
  for (int i = 0; i < 8; ++i) {
    a[i]     = p[kh + i];
    a[i + 8] = p[kh + 8 + i];
  }
  return a;
}
// B-matrix 32x16 bf16 fragment: lane holds column N=lane&15,
// K = k0 + 16*(lane>>4) + i (contiguous 16).
__device__ __forceinline__ v16bf fragB_gbf(const __bf16* m, int stride, int row,
                                           int k0, int klim, int lane) {
  const __bf16* p = m + (size_t)row * stride;
  const __bf16 z = __builtin_bit_cast(__bf16, (unsigned short)0);
  int kh = k0 + ((lane >> 4) << 4);
  v16bf b;
#pragma unroll
  for (int i = 0; i < 16; ++i) {
    int k = kh + i;
    b[i] = (k < klim) ? p[k] : z;
  }
  return b;
}
__device__ __forceinline__ v16bf fragB_lds(const __bf16* m, int stride, int row,
                                           int k0, int lane) {
  const __bf16* p = m + (size_t)row * stride;
  int kh = k0 + ((lane >> 4) << 4);
  v16bf b;
#pragma unroll
  for (int i = 0; i < 16; ++i) b[i] = p[kh + i];
  return b;
}

// ---------- Tensor Data Mover: 2D tile (bf16) -> LDS ----------
__device__ __forceinline__ void tdm_load(v4u g0, v8i g1) {
  v4i z4 = {0, 0, 0, 0};
#if defined(__clang_major__) && __clang_major__ >= 23
  v8i z8 = {0, 0, 0, 0, 0, 0, 0, 0};
  __builtin_amdgcn_tensor_load_to_lds(g0, g1, z4, z4, z8, 0);
#else
  __builtin_amdgcn_tensor_load_to_lds(g0, g1, z4, z4, 0);
#endif
}
// Load W_fc1 k-chunk: 256 rows x 32 cols bf16, row stride 1024 elements.
__device__ __forceinline__ void tdm_issue_wt(const __bf16* gsrc, unsigned lds_off) {
  unsigned long long ga = (unsigned long long)(size_t)gsrc;
  v4u g0;
  g0.x = 1u;                                                  // count=1 (valid)
  g0.y = lds_off;                                             // lds_addr
  g0.z = (unsigned)ga;                                        // global_addr[31:0]
  g0.w = (unsigned)((ga >> 32) & 0x1FFFFFFull) | (2u << 30);  // addr[56:32] | type=2
  v8i g1;
  g1[0] = (int)(1u << 16);      // workgroup_mask=0 | data_size=1 (2 bytes)
  g1[1] = (int)(1024u << 16);   // tensor_dim0[15:0]=1024 at bits 63:48
  g1[2] = (int)(256u << 16);    // dim0[31:16]=0 | tensor_dim1[15:0]=256
  g1[3] = (int)(32u << 16);     // dim1[31:16]=0 | tile_dim0=32
  g1[4] = 256;                  // tile_dim1=256 | tile_dim2=0
  g1[5] = 1024;                 // tensor_dim0_stride[31:0]=1024
  g1[6] = 0;                    // stride[47:32]=0 | dim1_stride[15:0]=0
  g1[7] = 0;                    // dim1_stride[47:16]=0
  tdm_load(g0, g1);
}

// ======================= small prep kernels =======================
__global__ void cvt_kernel(const float* __restrict__ s, __bf16* __restrict__ d, int n) {
  int i = blockIdx.x * blockDim.x + threadIdx.x;
  if (i < n) d[i] = b16(s[i]);
}
__global__ void copy_kernel(const float* __restrict__ s, float* __restrict__ d, int n) {
  int i = blockIdx.x * blockDim.x + threadIdx.x;
  if (i < n) d[i] = s[i];
}
__global__ void mask_init_kernel(const float* __restrict__ avail, float* __restrict__ mask0) {
  int i = blockIdx.x * blockDim.x + threadIdx.x;
  if (i >= B_ * NP1) return;
  int b = i / NP1, n = i % NP1;
  mask0[i] = (n < N_) ? avail[(size_t)b * N_ + n] : 1.0f;
}
__global__ void mask_force_kernel(const int* __restrict__ sel, float* __restrict__ mask0) {
  int i = blockIdx.x * blockDim.x + threadIdx.x;
  if (i >= B_ * S_) return;
  int b = i >> 5, t = i & 31;
  mask0[(size_t)b * NP1 + sel[b * S_ + t]] = 1.0f;
}
__global__ void pk_end_kernel(const float* __restrict__ endv, float* __restrict__ pkf,
                              __bf16* __restrict__ pkb) {
  int i = blockIdx.x * blockDim.x + threadIdx.x;
  if (i >= B_ * D_KEY) return;
  int b = i >> 5, k = i & 31;
  size_t o = ((size_t)b * NP1 + N_) * D_KEY + k;
  float v = endv[k];
  pkf[o] = v;
  pkb[o] = b16(v);
}

// ============ pad_key GEMM: (B*N,256) x (256->32), WMMA bf16 ============
__global__ void __launch_bounds__(256) key_gemm(const float* __restrict__ ent,
                                                const __bf16* __restrict__ Wkeyb,
                                                const float* __restrict__ b_key,
                                                float* __restrict__ pkf,
                                                __bf16* __restrict__ pkb) {
  int lane = threadIdx.x & 31, wv = threadIdx.x >> 5;
  int tile = blockIdx.x * 8 + wv;          // 8192 mtiles x 2 ntiles
  int mt = tile >> 1, nt = tile & 1;
  int row = lane & 15, rb = (lane >> 4) << 3;
  v8f acc = {};
  int erow = mt * 16 + row;
  for (int ks = 0; ks < D_ENT / 32; ++ks) {
    v16bf a = fragA_g32(ent, D_ENT, erow, ks * 32, D_ENT, lane);
    v16bf b = fragB_gbf(Wkeyb, D_ENT, nt * 16 + row, ks * 32, D_ENT, lane);
    acc = wmma_bf16(a, b, acc);
  }
  int col = nt * 16 + row;
  float bias = b_key[col];
#pragma unroll
  for (int r = 0; r < 8; ++r) {
    int e = mt * 16 + rb + r;               // entity row 0..131071
    int b = e >> 9, n = e & 511;
    size_t o = ((size_t)b * NP1 + n) * D_KEY + col;
    float v = acc[r] + bias;
    pkf[o] = v;
    pkb[o] = b16(v);
  }
}

// ============ func_embed: relu(mask_type @ Wfunc^T + b_func) + b_fc1 ============
__global__ void __launch_bounds__(256) func_gemm(const float* __restrict__ tmask,
                                                 const __bf16* __restrict__ Wfuncb,
                                                 const float* __restrict__ b_func,
                                                 const float* __restrict__ b_fc1,
                                                 float* __restrict__ fe) {
  int lane = threadIdx.x & 31, wv = threadIdx.x >> 5;
  int tile = blockIdx.x * 8 + wv;          // 16 mtiles x 16 ntiles
  int mt = tile >> 4, nt = tile & 15;
  int row = lane & 15, rb = (lane >> 4) << 3;
  v8f acc = {};
  for (int ks = 0; ks < 9; ++ks) {         // K = 259 padded to 288
    v16bf a = fragA_g32(tmask, D_TYPE, mt * 16 + row, ks * 32, D_TYPE, lane);
    v16bf b = fragB_gbf(Wfuncb, D_TYPE, nt * 16 + row, ks * 32, D_TYPE, lane);
    acc = wmma_bf16(a, b, acc);
  }
  int col = nt * 16 + row;
  float bias = b_func[col], bias1 = b_fc1[col];
#pragma unroll
  for (int r = 0; r < 8; ++r) {
    int m = mt * 16 + rb + r;
    float v = acc[r] + bias;
    fe[(size_t)m * D_FUNC + col] = (v > 0.0f ? v : 0.0f) + bias1;
  }
}

// ============ persistent scan: 16 WGs x 16 batch rows, 33 steps ============
__global__ void __launch_bounds__(256) scan_kernel(
    float* __restrict__ embW, const float* __restrict__ fe,
    const __bf16* __restrict__ Wfc1b, const __bf16* __restrict__ Wfc2b,
    const __bf16* __restrict__ Wihb, const __bf16* __restrict__ Whhb,
    const __bf16* __restrict__ Wembb, const float* __restrict__ b_fc2,
    const float* __restrict__ b_ih, const float* __restrict__ b_hh,
    const float* __restrict__ b_embed, const float* __restrict__ pkf,
    const int* __restrict__ sel, float* __restrict__ Hout,
    float* __restrict__ out_emb) {
  __shared__ __bf16 s_wt[2][256 * 32];   // 2x16KB: TDM-staged W_fc1 k-chunk
  __shared__ __bf16 s_embC[2][16 * 32];  // 2x1KB: emb k-chunk (f32->bf16)
  __shared__ __bf16 s_y[16 * D_FUNC];    // fc1 output (relu'd)
  __shared__ __bf16 s_x[16 * D_KEY];     // fc2 output
  __shared__ __bf16 s_hb[16 * H_];       // h as bf16
  __shared__ __bf16 s_sel[16 * D_KEY];   // gathered selected keys
  __shared__ float  s_g[16 * 128];       // LSTM gates
  __shared__ float  s_c[16 * H_];

  const int tid = threadIdx.x;
  const int lane = tid & 31, wv = tid >> 5;
  const int bg = blockIdx.x;              // batch rows bg*16 .. bg*16+15
  const int row = lane & 15, rb = (lane >> 4) << 3;
  const unsigned wt_lds[2] = {(unsigned)(size_t)&s_wt[0][0],
                              (unsigned)(size_t)&s_wt[1][0]};

  for (int e = tid; e < 16 * H_; e += 256) { s_c[e] = 0.0f; s_hb[e] = b16(0.0f); }
  __syncthreads();

  for (int t = 0; t < SP1; ++t) {
    // ---- fc1: y = relu(emb @ Wfc1^T + b_fc1 + func_embed) ----
    // K split into 32 chunks of 32; W_fc1 chunk DMA'd by the TDM (double-buffered),
    // emb chunk staged f32->bf16 by all threads.
    {
      v8f acc0 = {}, acc1 = {};
      const int nt0 = wv * 2, nt1 = wv * 2 + 1;
      if (wv == 0) tdm_issue_wt(Wfc1b, wt_lds[0]);
      for (int e = tid; e < 16 * 32; e += 256) {
        int m = e >> 5, j = e & 31;
        s_embC[0][e] = b16(embW[(size_t)(bg * 16 + m) * D_IN + j]);
      }
      __syncthreads();
      for (int ks = 0; ks < 32; ++ks) {
        const int cur = ks & 1, nxt = cur ^ 1;
        if (ks + 1 < 32) {
          if (wv == 0) tdm_issue_wt(Wfc1b + (ks + 1) * 32, wt_lds[nxt]);
          for (int e = tid; e < 16 * 32; e += 256) {
            int m = e >> 5, j = e & 31;
            s_embC[nxt][e] = b16(embW[(size_t)(bg * 16 + m) * D_IN + (ks + 1) * 32 + j]);
          }
          if (wv == 0) __builtin_amdgcn_s_wait_tensorcnt(1);   // chunk ks landed
        } else {
          if (wv == 0) __builtin_amdgcn_s_wait_tensorcnt(0);
        }
        __syncthreads();                    // publish wt[cur] + embC
        v16bf a  = fragA_lds(s_embC[cur], 32, row, 0, lane);
        v16bf b0 = fragB_lds(s_wt[cur], 32, nt0 * 16 + row, 0, lane);
        v16bf b1 = fragB_lds(s_wt[cur], 32, nt1 * 16 + row, 0, lane);
        acc0 = wmma_bf16(a, b0, acc0);
        acc1 = wmma_bf16(a, b1, acc1);
        __syncthreads();                    // release cur before next overwrite
      }
#pragma unroll
      for (int half = 0; half < 2; ++half) {
        v8f acc = half ? acc1 : acc0;
        int col = (half ? nt1 : nt0) * 16 + row;
#pragma unroll
        for (int r = 0; r < 8; ++r) {
          int m = rb + r;
          float v = acc[r] + fe[(size_t)(bg * 16 + m) * D_FUNC + col];
          s_y[m * D_FUNC + col] = b16(v > 0.0f ? v : 0.0f);
        }
      }
    }
    __syncthreads();

    // ---- fc2: x = y @ Wfc2^T + b_fc2 (2 tiles, waves 0-1) ----
    if (wv < 2) {
      v8f acc = {};
      for (int ks = 0; ks < D_FUNC / 32; ++ks) {
        v16bf a = fragA_lds(s_y, D_FUNC, row, ks * 32, lane);
        v16bf b = fragB_gbf(Wfc2b, D_FUNC, wv * 16 + row, ks * 32, D_FUNC, lane);
        acc = wmma_bf16(a, b, acc);
      }
      int col = wv * 16 + row;
#pragma unroll
      for (int r = 0; r < 8; ++r) s_x[(rb + r) * D_KEY + col] = b16(acc[r] + b_fc2[col]);
    }
    __syncthreads();

    // ---- gates = x @ Wih^T + h @ Whh^T + b_ih + b_hh (8 tiles, 1/wave) ----
    {
      v16bf ax = fragA_lds(s_x, D_KEY, row, 0, lane);
      v16bf ah = fragA_lds(s_hb, H_, row, 0, lane);
      v16bf bi = fragB_gbf(Wihb, H_, wv * 16 + row, 0, H_, lane);
      v16bf bh = fragB_gbf(Whhb, H_, wv * 16 + row, 0, H_, lane);
      v8f acc = {};
      acc = wmma_bf16(ax, bi, acc);
      acc = wmma_bf16(ah, bh, acc);
      int col = wv * 16 + row;
      float bias = b_ih[col] + b_hh[col];
#pragma unroll
      for (int r = 0; r < 8; ++r) s_g[(rb + r) * 128 + col] = acc[r] + bias;
    }
    __syncthreads();

    // ---- LSTM pointwise; record h into Hout ----
    for (int e = tid; e < 16 * H_; e += 256) {
      int m = e >> 5, j = e & 31;
      float ig = s_g[m * 128 + j];
      float fg = s_g[m * 128 + 32 + j];
      float gg = s_g[m * 128 + 64 + j];
      float og = s_g[m * 128 + 96 + j];
      float c = sigf(fg) * s_c[e] + sigf(ig) * tanhf(gg);
      float h = sigf(og) * tanhf(c);
      s_c[e] = c;
      s_hb[e] = b16(h);
      Hout[((size_t)(bg * 16 + m) * SP1 + t) * H_ + j] = h;
    }
    __syncthreads();

    // ---- emb += (pad_key[b, sel[t]] / N) @ Wembed^T + b_embed  (t < 32) ----
    if (t < S_) {
      for (int e = tid; e < 16 * H_; e += 256) {
        int m = e >> 5, j = e & 31;
        int b = bg * 16 + m;
        int n = sel[b * S_ + t];
        s_sel[e] = b16(pkf[((size_t)b * NP1 + n) * D_KEY + j] * (1.0f / (float)N_));
      }
      __syncthreads();
      v16bf a = fragA_lds(s_sel, D_KEY, row, 0, lane);
      for (int q = 0; q < 8; ++q) {         // 64 ntiles / 8 waves
        int nt = wv * 8 + q;
        v16bf b = fragB_gbf(Wembb, D_KEY, nt * 16 + row, 0, D_KEY, lane);
        v8f acc = {};
        acc = wmma_bf16(a, b, acc);
        int col = nt * 16 + row;
        float bias = b_embed[col];
#pragma unroll
        for (int r = 0; r < 8; ++r)
          embW[(size_t)(bg * 16 + rb + r) * D_IN + col] += acc[r] + bias;
      }
    }
    __syncthreads();
  }

  for (int e = tid; e < 16 * D_IN; e += 256)
    out_emb[(size_t)bg * 16 * D_IN + e] = embW[(size_t)bg * 16 * D_IN + e];
}

// ============ logits = H(33x32) x PK^T(32x513) per batch, + mask0 penalty ============
__global__ void __launch_bounds__(256) logits_gemm(const float* __restrict__ Hmat,
                                                   const __bf16* __restrict__ pkb,
                                                   const float* __restrict__ mask0,
                                                   float* __restrict__ out) {
  int lane = threadIdx.x & 31, wv = threadIdx.x >> 5;
  int tile = blockIdx.x * 8 + wv;        // B * (3 mtiles * 33 ntiles)
  if (tile >= B_ * 99) return;
  int b = tile / 99, rem = tile % 99;
  int mt = rem / 33, nt = rem % 33;
  int row = lane & 15, rb = (lane >> 4) << 3;
  int s = mt * 16 + row;
  int n = nt * 16 + row;
  v16bf a  = fragA_g32(Hmat + (size_t)b * SP1 * H_, H_, s, 0, (s < SP1) ? H_ : 0, lane);
  v16bf bb = fragB_gbf(pkb + (size_t)b * NP1 * D_KEY, D_KEY, n, 0, (n < NP1) ? H_ : 0, lane);
  v8f acc = {};
  acc = wmma_bf16(a, bb, acc);
  int col = nt * 16 + row;
  if (col < NP1) {
    float pen = (1.0f - mask0[(size_t)b * NP1 + col]) * 1e9f;
#pragma unroll
    for (int r = 0; r < 8; ++r) {
      int ss = mt * 16 + rb + r;
      if (ss < SP1) out[((size_t)b * SP1 + ss) * NP1 + col] = acc[r] - pen;
    }
  }
}

// fixup: unit selected at step t becomes masked (-1e9) for steps > t (first occurrence only)
__global__ void fixup_kernel(const int* __restrict__ sel, float* __restrict__ out) {
  int i = blockIdx.x * blockDim.x + threadIdx.x;
  if (i >= B_ * S_) return;
  int b = i >> 5, t = i & 31;
  int n = sel[b * S_ + t];
  for (int tt = 0; tt < t; ++tt)
    if (sel[b * S_ + tt] == n) return;     // not the first occurrence
  for (int s = t + 1; s < SP1; ++s)
    out[((size_t)b * SP1 + s) * NP1 + n] -= 1e9f;
}

// ======================= host launcher =======================
extern "C" void kernel_launch(void* const* d_in, const int* in_sizes, int n_in,
                              void* d_out, int out_size, void* d_ws, size_t ws_size,
                              hipStream_t stream) {
  (void)in_sizes; (void)n_in; (void)out_size; (void)ws_size;
  const float* embedding = (const float*)d_in[0];
  const float* tmask = (const float*)d_in[1];
  const float* avail = (const float*)d_in[2];
  const float* ent   = (const float*)d_in[3];
  const int*   sel   = (const int*)d_in[4];
  const float* endv  = (const float*)d_in[5];
  const float* W_key = (const float*)d_in[6];  const float* b_key  = (const float*)d_in[7];
  const float* W_func= (const float*)d_in[8];  const float* b_func = (const float*)d_in[9];
  const float* W_fc1 = (const float*)d_in[10]; const float* b_fc1  = (const float*)d_in[11];
  const float* W_fc2 = (const float*)d_in[12]; const float* b_fc2  = (const float*)d_in[13];
  const float* W_emb = (const float*)d_in[14]; const float* b_emb  = (const float*)d_in[15];
  const float* W_ih  = (const float*)d_in[16]; const float* b_ih   = (const float*)d_in[17];
  const float* W_hh  = (const float*)d_in[18]; const float* b_hh   = (const float*)d_in[19];

  char* base = (char*)d_ws;
  size_t off = 0;
  auto alloc = [&](size_t bytes) {
    char* p = base + off;
    off = (off + bytes + 255) & ~(size_t)255;
    return p;
  };
  float*  pkf    = (float*) alloc((size_t)B_ * NP1 * D_KEY * 4);
  __bf16* pkb    = (__bf16*)alloc((size_t)B_ * NP1 * D_KEY * 2);
  float*  mask0  = (float*) alloc((size_t)B_ * NP1 * 4);
  float*  fe     = (float*) alloc((size_t)B_ * D_FUNC * 4);
  float*  Hm     = (float*) alloc((size_t)B_ * SP1 * H_ * 4);
  float*  embW   = (float*) alloc((size_t)B_ * D_IN * 4);
  __bf16* Wfc1b  = (__bf16*)alloc((size_t)D_FUNC * D_IN * 2);
  __bf16* Wfc2b  = (__bf16*)alloc((size_t)D_KEY * D_FUNC * 2);
  __bf16* Wkeyb  = (__bf16*)alloc((size_t)D_KEY * D_ENT * 2);
  __bf16* Wfuncb = (__bf16*)alloc((size_t)D_FUNC * D_TYPE * 2);
  __bf16* Wembb  = (__bf16*)alloc((size_t)D_IN * D_KEY * 2);
  __bf16* Wihb   = (__bf16*)alloc((size_t)128 * H_ * 2);
  __bf16* Whhb   = (__bf16*)alloc((size_t)128 * H_ * 2);

  auto g = [](int n) { return (n + 255) / 256; };

  cvt_kernel<<<g(D_FUNC * D_IN), 256, 0, stream>>>(W_fc1, Wfc1b, D_FUNC * D_IN);
  cvt_kernel<<<g(D_KEY * D_FUNC), 256, 0, stream>>>(W_fc2, Wfc2b, D_KEY * D_FUNC);
  cvt_kernel<<<g(D_KEY * D_ENT), 256, 0, stream>>>(W_key, Wkeyb, D_KEY * D_ENT);
  cvt_kernel<<<g(D_FUNC * D_TYPE), 256, 0, stream>>>(W_func, Wfuncb, D_FUNC * D_TYPE);
  cvt_kernel<<<g(D_IN * D_KEY), 256, 0, stream>>>(W_emb, Wembb, D_IN * D_KEY);
  cvt_kernel<<<g(128 * H_), 256, 0, stream>>>(W_ih, Wihb, 128 * H_);
  cvt_kernel<<<g(128 * H_), 256, 0, stream>>>(W_hh, Whhb, 128 * H_);
  copy_kernel<<<g(B_ * D_IN), 256, 0, stream>>>(embedding, embW, B_ * D_IN);
  mask_init_kernel<<<g(B_ * NP1), 256, 0, stream>>>(avail, mask0);
  mask_force_kernel<<<g(B_ * S_), 256, 0, stream>>>(sel, mask0);
  pk_end_kernel<<<g(B_ * D_KEY), 256, 0, stream>>>(endv, pkf, pkb);

  key_gemm<<<(B_ * N_ / 16 * 2) / 8, 256, 0, stream>>>(ent, Wkeyb, b_key, pkf, pkb);
  func_gemm<<<(16 * 16) / 8, 256, 0, stream>>>(tmask, Wfuncb, b_func, b_fc1, fe);

  scan_kernel<<<B_ / 16, 256, 0, stream>>>(embW, fe, Wfc1b, Wfc2b, Wihb, Whhb, Wembb,
                                           b_fc2, b_ih, b_hh, b_emb, pkf, sel, Hm,
                                           (float*)d_out + (size_t)B_ * SP1 * NP1);

  logits_gemm<<<(B_ * 99 + 7) / 8, 256, 0, stream>>>(Hm, pkb, mask0, (float*)d_out);
  fixup_kernel<<<g(B_ * S_), 256, 0, stream>>>(sel, (float*)d_out);
}
